// Decoder_19189913878739
// MI455X (gfx1250) — matrix operand order repaired and checked by reading, not compiled, not run on hardware
//
#include <hip/hip_runtime.h>
#include <hip/hip_bf16.h>

// ---- CDNA5 WMMA types ----
typedef __attribute__((ext_vector_type(16))) __bf16 v16bf;
typedef __attribute__((ext_vector_type(8)))  float  v8f;

// Sizes (fixed by the reference)
#define B_TOT 1024
#define LD    256
#define H     512
#define H3    1536
#define T_STEPS 128

#define WMMA_BF16(A, B, C) \
    __builtin_amdgcn_wmma_f32_16x16x32_bf16(false, (A), false, (B), (short)0, (C), false, false)

__device__ __forceinline__ unsigned short f2bf(float f) {
    unsigned int u = __float_as_uint(f);
    unsigned int r = u + 0x7FFFu + ((u >> 16) & 1u);   // round-to-nearest-even
    return (unsigned short)(r >> 16);
}

// -------------------------------------------------------------------------
// Kernel 1: per-row setup.  h0 = tanh(latent@Wd + bd); x = latent@W + b0,
// with b_rec folded in for the z/r gates (h-gate recurrent bias must stay
// separate because hh = tanh(xh + r*(h@U_h + b_rec_h))).
// -------------------------------------------------------------------------
__global__ __launch_bounds__(256) void setup_proj(
        const float* __restrict__ latent, const float* __restrict__ Wd,
        const float* __restrict__ bd,     const float* __restrict__ W,
        const float* __restrict__ b,
        float* __restrict__ h0_ws, float* __restrict__ x_ws) {
    __shared__ float lrow[LD];
    const int row = blockIdx.x;
    const int tid = threadIdx.x;
    lrow[tid] = latent[row * LD + tid];
    __syncthreads();

    #pragma unroll
    for (int j = 0; j < 2; ++j) {
        const int c = tid + j * 256;
        float acc = bd[c];
        for (int k = 0; k < LD; ++k) acc += lrow[k] * Wd[k * H + c];
        h0_ws[row * H + c] = tanhf(acc);
    }
    #pragma unroll
    for (int j = 0; j < 6; ++j) {
        const int c = tid + j * 256;
        float acc = b[c];                                // input bias b[0]
        for (int k = 0; k < LD; ++k) acc += lrow[k] * W[k * H3 + c];
        if (c < 2 * H) acc += b[H3 + c];                 // fold b_rec for z,r
        x_ws[row * H3 + c] = acc;
    }
}

// -------------------------------------------------------------------------
// Kernel 2: pack U (f32 [512][1536]) into bf16 WMMA B-fragment layout.
// Tile (kb, ct): 32 lanes x 16 bf16 = 1 KB contiguous.  tileIdx = ct*16 + kb
// so the K-loop walks contiguous memory.  B layout (16x16x32 bf16):
//   lanes 0-15: N=lane, K={0..7,16..23} ; lanes 16-31: N=lane-16, K={8..15,24..31}.
// -------------------------------------------------------------------------
__global__ __launch_bounds__(256) void pack_u(
        const float* __restrict__ U, unsigned short* __restrict__ upack) {
    const int idx = blockIdx.x * blockDim.x + threadIdx.x;   // < 512*1536
    const int k = idx / H3, c = idx % H3;
    const int kb = k >> 5, ko = k & 31;
    const int ct = c >> 4, n  = c & 15;
    const int lane = n + (((ko >> 3) & 1) ? 16 : 0);
    const int slot = (ko & 7) + 8 * (ko >> 4);
    upack[(((ct * 16 + kb) * 32) + lane) * 16 + slot] = f2bf(U[idx]);
}

// -------------------------------------------------------------------------
// Kernel 3: persistent GRU recurrence.  One block = 16 batch rows, 16 waves;
// wave w owns h columns [32w, 32w+32) = 2 tiles in each of the 3 gates, so
// the 128-step recurrence needs no inter-block synchronization.
//
// LDS (dynamic, 114688 B):
//   hpack: 16 KB, h in packed A-fragment layout [kb][lane][16 bf16]
//          (A 16x32 bf16: lanes 0-15: M=lane, K={0..7,16..23};
//           lanes 16-31: M=lane-16, K={8..15,24..31}); wave w owns kb==w.
//   xpack: 96 KB, acc-init / x_h in C-fragment layout [waveTile][lane] v8f.
// -------------------------------------------------------------------------
__global__ __launch_bounds__(512, 1) void gru_recurrence(
        const unsigned short* __restrict__ upack,
        const float* __restrict__ h0_ws,
        const float* __restrict__ x_ws,
        const float* __restrict__ b,
        float* __restrict__ out) {
    extern __shared__ char smem[];
    v16bf* hpack = reinterpret_cast<v16bf*>(smem);             // 16*32 frags
    v8f*   xpack = reinterpret_cast<v8f*>(smem + 16384);       // 96*32 v8f
    unsigned short* hb = reinterpret_cast<unsigned short*>(smem);

    const int w     = threadIdx.x >> 5;              // wave 0..15
    const int lane  = threadIdx.x & 31;
    const int n     = lane & 15;                     // tile column
    const int mbase = (lane < 16) ? 0 : 8;           // C/D layout row base
    const int row0  = blockIdx.x * 16;
    const int cbase = w * 32;

    const v16bf* Up = reinterpret_cast<const v16bf*>(upack);

    float hreg[2][8];                                // h at this lane's (m,n)
    float brh[2];                                    // b_rec_h broadcast

    // ---- init: h0 -> regs + hpack ; x -> xpack (C-fragment layout) ----
    #pragma unroll
    for (int tt = 0; tt < 2; ++tt) {
        const int col = cbase + 16 * tt + n;
        brh[tt] = b[H3 + 2 * H + col];
        #pragma unroll
        for (int g = 0; g < 3; ++g) {
            v8f v;
            #pragma unroll
            for (int i = 0; i < 8; ++i)
                v[i] = x_ws[(size_t)(row0 + mbase + i) * H3 + g * H + col];
            xpack[(w * 6 + g * 2 + tt) * 32 + lane] = v;
        }
        #pragma unroll
        for (int i = 0; i < 8; ++i)
            hreg[tt][i] = h0_ws[(size_t)(row0 + mbase + i) * H + col];
    }
    // hpack write base: element (m = mbase+i, ko = 16*tt + n) of block kb=w
    const int wbase = (w * 32 + mbase + ((n >= 8) ? 16 : 0)) * 16 + (n & 7);
    #pragma unroll
    for (int tt = 0; tt < 2; ++tt)
        #pragma unroll
        for (int i = 0; i < 8; ++i)
            hb[wbase + i * 16 + 8 * tt] = f2bf(hreg[tt][i]);
    __syncthreads();

    // per-tile U fragment base pointers: tileIdx = (g*32 + 2w + tt)*16 + kb
    const v16bf* up0 = Up + (( 0 + 2 * w + 0) * 16) * 32 + lane;
    const v16bf* up1 = Up + (( 0 + 2 * w + 1) * 16) * 32 + lane;
    const v16bf* up2 = Up + ((32 + 2 * w + 0) * 16) * 32 + lane;
    const v16bf* up3 = Up + ((32 + 2 * w + 1) * 16) * 32 + lane;
    const v16bf* up4 = Up + ((64 + 2 * w + 0) * 16) * 32 + lane;
    const v16bf* up5 = Up + ((64 + 2 * w + 1) * 16) * 32 + lane;

    for (int t = 0; t < T_STEPS; ++t) {
        // accumulator init: z/r from xpack (x + b_rec), h-gate from b_rec_h
        v8f az0 = xpack[(w * 6 + 0) * 32 + lane];
        v8f az1 = xpack[(w * 6 + 1) * 32 + lane];
        v8f ar0 = xpack[(w * 6 + 2) * 32 + lane];
        v8f ar1 = xpack[(w * 6 + 3) * 32 + lane];
        v8f ah0, ah1;
        #pragma unroll
        for (int i = 0; i < 8; ++i) { ah0[i] = brh[0]; ah1[i] = brh[1]; }

        // K loop: rolled (unroll 1) so U loads stay inside and cannot be
        // hoisted/spilled; two fragment sets per iteration so the second
        // set's loads overlap the first set's WMMAs.
        #pragma unroll 1
        for (int kb = 0; kb < 16; kb += 2) {
            const int k0 = kb * 32, k1 = (kb + 1) * 32;
            v16bf aA = hpack[k0 + lane];
            v16bf bA0 = up0[k0], bA1 = up1[k0], bA2 = up2[k0],
                  bA3 = up3[k0], bA4 = up4[k0], bA5 = up5[k0];
            v16bf aB = hpack[k1 + lane];
            v16bf bB0 = up0[k1], bB1 = up1[k1], bB2 = up2[k1],
                  bB3 = up3[k1], bB4 = up4[k1], bB5 = up5[k1];

            az0 = WMMA_BF16(aA, bA0, az0);
            az1 = WMMA_BF16(aA, bA1, az1);
            ar0 = WMMA_BF16(aA, bA2, ar0);
            ar1 = WMMA_BF16(aA, bA3, ar1);
            ah0 = WMMA_BF16(aA, bA4, ah0);
            ah1 = WMMA_BF16(aA, bA5, ah1);

            az0 = WMMA_BF16(aB, bB0, az0);
            az1 = WMMA_BF16(aB, bB1, az1);
            ar0 = WMMA_BF16(aB, bB2, ar0);
            ar1 = WMMA_BF16(aB, bB3, ar1);
            ah0 = WMMA_BF16(aB, bB4, ah0);
            ah1 = WMMA_BF16(aB, bB5, ah1);
        }
        __syncthreads();                 // all waves done reading hpack

        // elementwise gate math + output store + h repack
        v8f xh0 = xpack[(w * 6 + 4) * 32 + lane];
        v8f xh1 = xpack[(w * 6 + 5) * 32 + lane];
        #pragma unroll
        for (int tt = 0; tt < 2; ++tt) {
            const v8f& azv = tt ? az1 : az0;
            const v8f& arv = tt ? ar1 : ar0;
            const v8f& ahv = tt ? ah1 : ah0;
            const v8f& xhv = tt ? xh1 : xh0;
            #pragma unroll
            for (int i = 0; i < 8; ++i) {
                const float z  = 1.0f / (1.0f + __expf(-azv[i]));
                const float rg = 1.0f / (1.0f + __expf(-arv[i]));
                const float hh = tanhf(xhv[i] + rg * ahv[i]);
                const float hn = z * hreg[tt][i] + (1.0f - z) * hh;
                hreg[tt][i] = hn;
                out[(size_t)(row0 + mbase + i) * (T_STEPS * H)
                    + (size_t)t * H + cbase + 16 * tt + n] = hn;
                hb[wbase + i * 16 + 8 * tt] = f2bf(hn);
            }
        }
        __syncthreads();                 // hpack updated for next step
    }
}

// -------------------------------------------------------------------------
extern "C" void kernel_launch(void* const* d_in, const int* in_sizes, int n_in,
                              void* d_out, int out_size, void* d_ws, size_t ws_size,
                              hipStream_t stream) {
    const float* latent = (const float*)d_in[0];   // [1024,256]
    const float* Wd     = (const float*)d_in[1];   // [256,512]
    const float* bd     = (const float*)d_in[2];   // [512]
    const float* W      = (const float*)d_in[3];   // [256,1536]
    const float* U      = (const float*)d_in[4];   // [512,1536]
    const float* b      = (const float*)d_in[5];   // [2,1536]
    float* out = (float*)d_out;

    // Workspace layout (~9.9 MB total)
    char* ws = (char*)d_ws;
    unsigned short* upack = (unsigned short*)ws;                       // 1.5 MB
    float* h0_ws = (float*)(ws + (size_t)H * H3 * sizeof(unsigned short));
    float* x_ws  = (float*)((char*)h0_ws + (size_t)B_TOT * H * sizeof(float));

    setup_proj<<<B_TOT, 256, 0, stream>>>(latent, Wd, bd, W, b, h0_ws, x_ws);
    pack_u<<<(H * H3) / 256, 256, 0, stream>>>(U, upack);

    const size_t lds_bytes = 16384 + 96 * 32 * 32;   // hpack + xpack = 114688
    gru_recurrence<<<B_TOT / 16, 512, lds_bytes, stream>>>(
        upack, h0_ws, x_ws, b, out);
}
